// CAM_63548336112251
// MI455X (gfx1250) — matrix operand order repaired and checked by reading, not compiled, not run on hardware
//
#include <hip/hip_runtime.h>
#include <hip/hip_bf16.h>
#include <math.h>

// ---------------------------------------------------------------------------
// CAM few-shot attention module for MI455X (gfx1250, wave32, WMMA).
//
// Cosine-similarity GEMM (625x640)x(640x50000) as per-(p,q) 25x25 blocks via
// v_wmma_f32_16x16x32_f16. Each wave owns a full p: 2x2 16x16 tiles (4 f32
// accumulators), 80 WMMAs per p, epilogue fully intra-wave (wave32 LDS ops
// are in-order per wave) -> NO barriers inside the p loop.
// ---------------------------------------------------------------------------

typedef __attribute__((ext_vector_type(16))) _Float16 v16h;
typedef __attribute__((ext_vector_type(8)))  float    v8f;

#define PP   25     // way*shot
#define QQ   2000   // way*query
#define CC   640    // channels (K dim)
#define HWX  25     // 5x5 spatial
#define KS   20     // K steps: 640 / 32
#define FRAG 512    // halves per fragment: 32 lanes * 16 halves
#define BN_EPS 1e-5f

#define WMMA16(a, b, c) \
  __builtin_amdgcn_wmma_f32_16x16x32_f16(false, (a), false, (b), (short)0, (c), false, false)

// ---------------------------------------------------------------------------
// Kernel 1: L2-normalize ftrain per (p, spatial) over C, emit f16 A-fragments
// in the CDNA5 WMMA 16-bit A 16x32 lane layout (CDNA5 ISA 7.12.2):
//   lane = 16*hi + m (m = row), half h -> K = 32*ks + (h<8 ? h+8*hi : 8+h+8*hi)
// M padded 25 -> 32 (2 m-tiles), pad rows zero.  afrag: [p][mt][ks][lane][16]
// ---------------------------------------------------------------------------
__global__ __launch_bounds__(128) void cam_prep_train(
    const float* __restrict__ ftrain, _Float16* __restrict__ afrag) {
  __shared__ float part[25 * 5];
  __shared__ float rnorm[25];
  const int p = blockIdx.x;
  const int t = threadIdx.x;
  const float* F = ftrain + p * CC * HWX;

  if (t < 125) {
    const int hw = t % 25, seg = t / 25;
    float s = 0.f;
    for (int c = seg * 128; c < seg * 128 + 128; ++c) {
      const float v = F[c * 25 + hw];
      s += v * v;
    }
    part[hw * 5 + seg] = s;
  }
  __syncthreads();
  if (t < 25) {
    float s = 0.f;
    for (int seg = 0; seg < 5; ++seg) s += part[t * 5 + seg];
    rnorm[t] = 1.f / fmaxf(sqrtf(s), 1e-12f);
  }
  __syncthreads();

  _Float16* A = afrag + p * (2 * KS * FRAG);
  for (int idx = t; idx < 2 * KS * FRAG; idx += 128) {
    const int h    = idx & 15;
    const int lane = (idx >> 4) & 31;
    const int ks   = (idx >> 9) % KS;
    const int mt   = idx / (KS * FRAG);
    const int m    = mt * 16 + (lane & 15);
    const int hi   = lane >> 4;
    const int k    = ks * 32 + (h < 8 ? h + 8 * hi : 8 + h + 8 * hi);
    const float v  = (m < HWX) ? F[k * 25 + m] * rnorm[m] : 0.f;
    A[idx] = (_Float16)v;
  }
}

// ---------------------------------------------------------------------------
// Kernel 2: one block (4 waves) per test sample q.
//  - block-wide: normalize ftest[q], build B-fragments in LDS (40 KB),
//    B 32x16 layout: lane = 16*hi + n, half h -> K = 32*ks + (h<8 ? h+8*hi : 8+h+8*hi).
//  - p-loop: wave w handles p = w, w+4, ... : 2x2 WMMA tiles (80 WMMAs/p)
//    into a private LDS cos block, then the whole MLP+attention epilogue
//    intra-wave. Per-wave am/cls accumulators, combined once at the end.
// ---------------------------------------------------------------------------
__global__ __launch_bounds__(128) void cam_main(
    const float* __restrict__ ftest, const _Float16* __restrict__ afrag,
    const float* __restrict__ w1, const float* __restrict__ b1,
    const float* __restrict__ bn_g, const float* __restrict__ bn_b,
    const float* __restrict__ bn_m, const float* __restrict__ bn_v,
    const float* __restrict__ w2, const float* __restrict__ b2,
    float* __restrict__ att_partial, float* __restrict__ out_ftest,
    float* __restrict__ out_am, float* __restrict__ out_cls) {
  __shared__ alignas(32) _Float16 bfrag[2 * KS * FRAG];  // 40 KB
  __shared__ float cosw[4][32 * 33];                      // per-wave cos block
  __shared__ float part[25 * 5];
  __shared__ float rn[25];
  __shared__ float gw[4][32], gtw[4][32], h1aw[4][8], h1bw[4][8];
  __shared__ float kkw[4][32], ktvw[4][32], tsw[4][32];
  __shared__ float amw[4][32], clsw[4][8];
  __shared__ float amfin[25], clsfin[8];

  const int q = blockIdx.x;
  const int t = threadIdx.x;
  const float* F = ftest + (size_t)q * CC * HWX;

  // --- column L2 norms of ftest[q] (block-wide) ---
  if (t < 125) {
    const int hw = t % 25, seg = t / 25;
    float s = 0.f;
    for (int c = seg * 128; c < seg * 128 + 128; ++c) {
      const float v = F[c * 25 + hw];
      s += v * v;
    }
    part[hw * 5 + seg] = s;
  }
  __syncthreads();
  if (t < 25) {
    float s = 0.f;
    for (int seg = 0; seg < 5; ++seg) s += part[t * 5 + seg];
    rn[t] = 1.f / fmaxf(sqrtf(s), 1e-12f);
  }
  __syncthreads();

  // --- build B fragments in LDS: [nt][ks][lane][16] (block-wide) ---
  for (int idx = t; idx < 2 * KS * FRAG; idx += 128) {
    const int h    = idx & 15;
    const int lane = (idx >> 4) & 31;
    const int ks   = (idx >> 9) % KS;
    const int nt   = idx / (KS * FRAG);
    const int n    = nt * 16 + (lane & 15);
    const int k    = ks * 32 + (h < 8 ? h + 8 * (lane >> 4) : 8 + h + 8 * (lane >> 4));
    const float v  = (n < HWX) ? F[k * 25 + n] * rn[n] : 0.f;
    bfrag[idx] = (_Float16)v;
  }
  __syncthreads();

  const int lane = t & 31;
  const int wav  = t >> 5;
  float* cosb = cosw[wav];

  if (lane < 25) amw[wav][lane] = 0.f;
  if (lane < 8)  clsw[wav][lane] = 0.f;
  // (intra-wave init; read only by this wave until the final combine)

  const int nlo   = lane & 15;
  const int mhalf = 8 * (lane >> 4);

  for (int p = wav; p < PP; p += 4) {
    // ---- 25x25 cos block: 2x2 tiles x K=640, all in this wave ----
    v8f a00 = {0.f,0.f,0.f,0.f,0.f,0.f,0.f,0.f};
    v8f a01 = a00, a10 = a00, a11 = a00;
    const _Float16* Ab = afrag + ((size_t)p * 2 * KS) * FRAG + lane * 16;
    const _Float16* Bb = bfrag + lane * 16;
    #pragma unroll 4
    for (int ks = 0; ks < KS; ++ks) {
      v16h fa0 = *(const v16h*)(Ab + ks * FRAG);             // m-tile 0
      v16h fa1 = *(const v16h*)(Ab + (KS + ks) * FRAG);      // m-tile 1
      v16h fb0 = *(const v16h*)(Bb + ks * FRAG);             // n-tile 0
      v16h fb1 = *(const v16h*)(Bb + (KS + ks) * FRAG);      // n-tile 1
      a00 = WMMA16(fa0, fb0, a00);
      a01 = WMMA16(fa0, fb1, a01);
      a10 = WMMA16(fa1, fb0, a10);
      a11 = WMMA16(fa1, fb1, a11);
    }
    #pragma unroll
    for (int r = 0; r < 8; ++r) {
      cosb[(mhalf + r) * 33 + nlo]            = a00[r];
      cosb[(mhalf + r) * 33 + 16 + nlo]       = a01[r];
      cosb[(16 + mhalf + r) * 33 + nlo]       = a10[r];
      cosb[(16 + mhalf + r) * 33 + 16 + nlo]  = a11[r];
    }
    // ---- epilogue: entirely intra-wave (wave32 LDS ops are in-order) ----
    if (lane < 25) {  // pooled means, both sides
      float s = 0.f, st = 0.f;
      for (int j = 0; j < 25; ++j) s  += cosb[lane * 33 + j];
      for (int i = 0; i < 25; ++i) st += cosb[i * 33 + lane];
      gw[wav][lane]  = s  * (1.f / 25.f);
      gtw[wav][lane] = st * (1.f / 25.f);
    }
    if (lane < 10) {  // layer1 (25->5) + BN(eval) + relu, both sides
      const int o = (lane < 5) ? lane : lane - 5;
      const float* gg = (lane < 5) ? gw[wav] : gtw[wav];
      float z = b1[o];
      for (int i = 0; i < 25; ++i) z += gg[i] * w1[o * 25 + i];
      z = (z - bn_m[o]) * rsqrtf(bn_v[o] + BN_EPS) * bn_g[o] + bn_b[o];
      float* dst = (lane < 5) ? h1aw[wav] : h1bw[wav];
      dst[o] = fmaxf(z, 0.f);
    }
    if (lane < 25) {  // layer2 (5->25) + relu, both sides
      float z = b2[lane], zt = b2[lane];
      for (int o = 0; o < 5; ++o) {
        z  += h1aw[wav][o] * w2[lane * 5 + o];
        zt += h1bw[wav][o] * w2[lane * 5 + o];
      }
      kkw[wav][lane]  = fmaxf(z, 0.f);
      ktvw[wav][lane] = fmaxf(zt, 0.f);
    }
    if (lane < 25) {  // attention projections
      float s = 0.f, st = 0.f;
      for (int i = 0; i < 25; ++i) s  += kkw[wav][i]  * cosb[i * 33 + lane];
      for (int j = 0; j < 25; ++j) st += ktvw[wav][j] * cosb[lane * 33 + j];
      att_partial[((size_t)q * 25 + p) * 25 + lane] = s;  // train side, reduce over q later
      amw[wav][lane] += st;                               // test side: sum over p
      tsw[wav][lane] = st;
    }
    if (lane == 0) {  // class logit bucket for way(p)
      float s = 0.f;
      for (int i = 0; i < 25; ++i) s += tsw[wav][i];
      clsw[wav][p / 5] += s;
    }
  }
  __syncthreads();

  // ---- combine per-wave accumulators, finalize per-q outputs ----
  if (t < 25)
    amfin[t] = (amw[0][t] + amw[1][t] + amw[2][t] + amw[3][t]) * (1.f / 25.f);
  if (t < 5)
    clsfin[t] = (clsw[0][t] + clsw[1][t] + clsw[2][t] + clsw[3][t]) * (1.f / 125.f);
  __syncthreads();
  if (t == 0) {        // am softmax
    float mx = -1e30f;
    for (int i = 0; i < 25; ++i) if (amfin[i] > mx) mx = amfin[i];
    float s = 0.f;
    for (int i = 0; i < 25; ++i) { const float e = expf(amfin[i] - mx); amfin[i] = e; s += e; }
    const float inv = 1.f / s;
    for (int i = 0; i < 25; ++i) amfin[i] *= inv;
  } else if (t == 32) {  // cls softmax
    float mx = -1e30f;
    for (int w = 0; w < 5; ++w) if (clsfin[w] > mx) mx = clsfin[w];
    float s = 0.f;
    for (int w = 0; w < 5; ++w) { const float e = expf(clsfin[w] - mx); clsfin[w] = e; s += e; }
    const float inv = 1.f / s;
    for (int w = 0; w < 5; ++w) clsfin[w] *= inv;
  }
  __syncthreads();
  if (t < 25) out_am[q * 25 + t] = amfin[t];
  if (t >= 32 && t < 37) out_cls[q * 5 + (t - 32)] = clsfin[t - 32];
  for (int idx = t; idx < CC * HWX; idx += 128)
    out_ftest[(size_t)q * CC * HWX + idx] = F[idx] * (amfin[idx % 25] + 1.f);
}

// ---------------------------------------------------------------------------
// Kernel 3: reduce att_partial over q, softmax, scale ftrain.
// ---------------------------------------------------------------------------
__global__ __launch_bounds__(128) void cam_train_out(
    const float* __restrict__ ftrain, const float* __restrict__ att_partial,
    float* __restrict__ out_ftrain) {
  __shared__ float part[25 * 5];
  __shared__ float attv[25];
  const int p = blockIdx.x;
  const int t = threadIdx.x;

  if (t < 125) {
    const int j = t % 25, seg = t / 25;
    float s = 0.f;
    for (int q = seg * 400; q < seg * 400 + 400; ++q)
      s += att_partial[((size_t)q * 25 + p) * 25 + j];
    part[j * 5 + seg] = s;
  }
  __syncthreads();
  if (t < 25) {
    float s = 0.f;
    for (int seg = 0; seg < 5; ++seg) s += part[t * 5 + seg];
    attv[t] = s * (1.f / (float)QQ);
  }
  __syncthreads();
  if (t == 0) {
    float mx = -1e30f;
    for (int i = 0; i < 25; ++i) if (attv[i] > mx) mx = attv[i];
    float s = 0.f;
    for (int i = 0; i < 25; ++i) { const float e = expf(attv[i] - mx); attv[i] = e; s += e; }
    const float inv = 1.f / s;
    for (int i = 0; i < 25; ++i) attv[i] *= inv;
  }
  __syncthreads();
  for (int idx = t; idx < CC * HWX; idx += 128)
    out_ftrain[(size_t)p * CC * HWX + idx] =
        ftrain[(size_t)p * CC * HWX + idx] * (attv[idx % 25] + 1.f);
}

// ---------------------------------------------------------------------------
// Host-side launch.
// d_out (floats): ftrain_out[400000] | ftest_out[32000000] | am[50000] | cls[10000]
// d_ws: A-fragments f16 [1,024,000 B] @0 ; att_partial f32 [5 MB] @1MB
// ---------------------------------------------------------------------------
extern "C" void kernel_launch(void* const* d_in, const int* in_sizes, int n_in,
                              void* d_out, int out_size, void* d_ws, size_t ws_size,
                              hipStream_t stream) {
  (void)in_sizes; (void)n_in; (void)out_size; (void)ws_size;

  const float* ftrain  = (const float*)d_in[0];
  const float* ftest   = (const float*)d_in[1];
  const float* conv1_w = (const float*)d_in[6];
  const float* conv1_b = (const float*)d_in[7];
  const float* bn_g    = (const float*)d_in[8];
  const float* bn_b    = (const float*)d_in[9];
  const float* bn_m    = (const float*)d_in[10];
  const float* bn_v    = (const float*)d_in[11];
  const float* conv2_w = (const float*)d_in[12];
  const float* conv2_b = (const float*)d_in[13];

  float* out        = (float*)d_out;
  float* out_ftrain = out;
  float* out_ftest  = out + 400000;
  float* out_am     = out + 32400000;
  float* out_cls    = out + 32450000;

  _Float16* afrag    = (_Float16*)d_ws;
  float* att_partial = (float*)((char*)d_ws + (1u << 20));

  cam_prep_train<<<PP, 128, 0, stream>>>(ftrain, afrag);
  cam_main<<<QQ, 128, 0, stream>>>(ftest, afrag,
                                   conv1_w, conv1_b, bn_g, bn_b, bn_m, bn_v,
                                   conv2_w, conv2_b,
                                   att_partial, out_ftest, out_am, out_cls);
  cam_train_out<<<PP, 128, 0, stream>>>(ftrain, att_partial, out_ftrain);
}
